// SetResNet_12000138625067
// MI455X (gfx1250) — matrix operand order repaired and checked by reading, not compiled, not run on hardware
//
#include <hip/hip_runtime.h>
#include <hip/hip_bf16.h>
#include <math.h>

typedef __attribute__((ext_vector_type(16))) _Float16 v16h;
typedef __attribute__((ext_vector_type(8)))  float    v8f;

#define EDIM  96
#define HEADS 6
#define DHEAD 16
#define NROWS 4096
#define SROW  36   // padded LDS row stride (floats) to stagger banks

// ---------------------------------------------------------------------------
// Kernel 1: qkv = feats @ w_in^T + b_in  -> f16 Qh/Kh (row-major, Q pre-scaled
// by 1/sqrt(DH)) and f16 Vt (transposed [H][DHEAD][N] for WMMA B-operand loads)
// ---------------------------------------------------------------------------
__global__ __launch_bounds__(288)
void qkv_proj_kernel(const float* __restrict__ feats,
                     const float* __restrict__ w_in,
                     const float* __restrict__ b_in,
                     _Float16* __restrict__ Qh,
                     _Float16* __restrict__ Kh,
                     _Float16* __restrict__ Vt)
{
    __shared__ float frow[EDIM];
    const int n = blockIdx.x;
    const int j = threadIdx.x;                 // 0..287 output slot
    if (j < EDIM) frow[j] = feats[(size_t)n * EDIM + j];
    __syncthreads();

    float acc = b_in[j];
    const float* w = w_in + (size_t)j * EDIM;
    #pragma unroll 8
    for (int k = 0; k < EDIM; ++k) acc = fmaf(frow[k], w[k], acc);

    if (j < EDIM) {                            // Q (scaled)
        const int h = j >> 4, d = j & 15;
        Qh[((size_t)h * NROWS + n) * DHEAD + d] = (_Float16)(acc * 0.25f);
    } else if (j < 2 * EDIM) {                 // K
        const int jj = j - EDIM;
        const int h = jj >> 4, d = jj & 15;
        Kh[((size_t)h * NROWS + n) * DHEAD + d] = (_Float16)acc;
    } else {                                   // V transposed
        const int jj = j - 2 * EDIM;
        const int h = jj >> 4, d = jj & 15;
        Vt[((size_t)h * DHEAD + d) * NROWS + n] = (_Float16)acc;
    }
}

// ---------------------------------------------------------------------------
// Kernel 2: flash-style group-masked attention. One wave32 per (head, 16-row
// query tile). 32 keys/iteration: 2 score WMMAs + 1 PV WMMA. Softmax is done
// per-lane in A-layout after a single LDS transpose of the masked scores.
// ---------------------------------------------------------------------------
__global__ __launch_bounds__(32)
void attn_kernel(const _Float16* __restrict__ Qh,
                 const _Float16* __restrict__ Kh,
                 const _Float16* __restrict__ Vt,
                 const int* __restrict__ group,
                 float* __restrict__ attn_out)
{
    __shared__ float Slds[16 * SROW];          // masked score tile, padded rows

    const int lane  = threadIdx.x;
    const int hi    = lane >> 4;               // which 16-lane half
    const int l15   = lane & 15;
    const int head  = blockIdx.y;
    const int qbase = blockIdx.x * 16;

    // Q as WMMA A operand: lane holds row (qbase+l15); K-dim 16 padded to 32.
    v16h aQ = {};
    {
        const _Float16* qrow = Qh + ((size_t)head * NROWS + qbase + l15) * DHEAD;
        const int k0 = hi * 8;
        #pragma unroll
        for (int i = 0; i < 8; ++i) aQ[i] = qrow[k0 + i];
        // aQ[8..15] stay zero (K = 16..31 pad)
    }

    // Group ids of the 8 query rows this lane owns in C-layout (M = hi*8 + r)
    int gq[8];
    #pragma unroll
    for (int r = 0; r < 8; ++r) gq[r] = group[qbase + hi * 8 + r];

    // Per-lane softmax state for query row (qbase + l15); lanes l and l+16
    // are kept coherent via xor-16 shuffles.
    float m_row = -1e9f;
    float l_row = 0.0f;
    v8f acc = {};                               // O accumulator, C-layout

    for (int kb = 0; kb < NROWS; kb += 32) {
        // K rows as B operand: column j = kb+l15; half hi holds d=hi*8..hi*8+7
        // in elements 0..7, elements 8..15 are the zero pad (d=16..31).
        v16h bK0 = {}, bK1 = {};
        {
            const _Float16* kr0 = Kh + ((size_t)head * NROWS + kb + l15) * DHEAD;
            const _Float16* kr1 = kr0 + 16 * DHEAD;
            #pragma unroll
            for (int i = 0; i < 8; ++i) { bK0[i] = kr0[hi * 8 + i]; bK1[i] = kr1[hi * 8 + i]; }
        }
        v8f czero = {};
        v8f S0 = __builtin_amdgcn_wmma_f32_16x16x32_f16(
                     false, aQ, false, bK0, (short)0, czero, false, false);
        v8f S1 = __builtin_amdgcn_wmma_f32_16x16x32_f16(
                     false, aQ, false, bK1, (short)0, czero, false, false);

        const int g0 = group[kb + l15];         // group of key column (tile 0)
        const int g1 = group[kb + 16 + l15];    // group of key column (tile 1)

        // Mask in C-layout and transpose through LDS (row M = hi*8 + r)
        #pragma unroll
        for (int r = 0; r < 8; ++r) {
            const int row = hi * 8 + r;
            Slds[row * SROW + l15]      = (gq[r] == g0) ? S0[r] : -1e9f;
            Slds[row * SROW + 16 + l15] = (gq[r] == g1) ? S1[r] : -1e9f;
        }
        __syncthreads();

        // Read my row's K-chunk in A-fragment order
        float s[16];
        {
            const float* srow = &Slds[l15 * SROW];
            const int k0 = hi * 8;
            #pragma unroll
            for (int i = 0; i < 8; ++i) {
                s[i]     = srow[k0 + i];
                s[8 + i] = srow[k0 + 16 + i];
            }
        }
        __syncthreads();                        // Slds reusable next iteration

        // Per-lane online softmax over this half-row; merge halves via xor-16
        float rm = s[0];
        #pragma unroll
        for (int i = 1; i < 16; ++i) rm = fmaxf(rm, s[i]);
        rm = fmaxf(rm, __shfl_xor(rm, 16, 32));
        const float mn = fmaxf(m_row, rm);
        const float sc = __expf(m_row - mn);

        v16h aP;
        float ps = 0.0f;
        #pragma unroll
        for (int i = 0; i < 16; ++i) {
            const float p = __expf(s[i] - mn);
            ps += p;
            aP[i] = (_Float16)p;
        }
        ps += __shfl_xor(ps, 16, 32);
        l_row = l_row * sc + ps;
        m_row = mn;

        // Rescale C-layout accumulator: row M's scale lives in lane M
        #pragma unroll
        for (int r = 0; r < 8; ++r) acc[r] *= __shfl(sc, hi * 8 + r, 32);

        // V columns as B operand: lane holds column d=l15, keys hi*8+0..7 and 16+hi*8+0..7
        v16h bV;
        {
            const _Float16* vrow =
                Vt + ((size_t)head * DHEAD + l15) * NROWS + kb + hi * 8;
            #pragma unroll
            for (int i = 0; i < DHEAD; ++i) bV[i] = vrow[i < 8 ? i : i + 8];
        }
        acc = __builtin_amdgcn_wmma_f32_16x16x32_f16(
                  false, aP, false, bV, (short)0, acc, false, false);
    }

    // Normalize and scatter: element (M = hi*8+r, d = l15)
    #pragma unroll
    for (int r = 0; r < 8; ++r) {
        const float lr = __shfl(l_row, hi * 8 + r, 32);
        attn_out[(size_t)(qbase + hi * 8 + r) * EDIM + head * DHEAD + l15] =
            acc[r] / lr;
    }
}

// ---------------------------------------------------------------------------
// Kernel 3: fold out_proj + classifier into one vector:
//   wv[k] = sum_e w_cls[e] * w_out[e][k],  wv[96] = w_cls·b_out + b_cls
// ---------------------------------------------------------------------------
__global__ __launch_bounds__(128)
void fuse_cls_kernel(const float* __restrict__ w_out,
                     const float* __restrict__ b_out,
                     const float* __restrict__ w_cls,
                     const float* __restrict__ b_cls,
                     float* __restrict__ wv)
{
    const int k = threadIdx.x;
    if (k < EDIM) {
        float s = 0.0f;
        for (int e = 0; e < EDIM; ++e) s = fmaf(w_cls[e], w_out[(size_t)e * EDIM + k], s);
        wv[k] = s;
    }
    if (k == EDIM) {
        float c = b_cls[0];
        for (int e = 0; e < EDIM; ++e) c = fmaf(w_cls[e], b_out[e], c);
        wv[EDIM] = c;
    }
}

// ---------------------------------------------------------------------------
// Kernel 4: logits + sigmoid
// ---------------------------------------------------------------------------
__global__ __launch_bounds__(256)
void cls_kernel(const float* __restrict__ attn,
                const float* __restrict__ wv,
                float* __restrict__ out)
{
    const int n = blockIdx.x * blockDim.x + threadIdx.x;
    if (n >= NROWS) return;
    float x = wv[EDIM];
    const float* a = attn + (size_t)n * EDIM;
    #pragma unroll 8
    for (int k = 0; k < EDIM; ++k) x = fmaf(a[k], wv[k], x);
    out[n] = 1.0f / (1.0f + __expf(-x));
}

// ---------------------------------------------------------------------------
// Workspace layout (needs ~3.94 MB):
//   [0)                 Qh  f16 [H][N][16]   786432 B
//   [786432)            Kh  f16 [H][N][16]   786432 B
//   [1572864)           Vt  f16 [H][16][N]   786432 B
//   [2359296)           attn f32 [N][96]    1572864 B
//   [3932160)           wv  f32 [97]            388 B
// ---------------------------------------------------------------------------
extern "C" void kernel_launch(void* const* d_in, const int* in_sizes, int n_in,
                              void* d_out, int out_size, void* d_ws, size_t ws_size,
                              hipStream_t stream) {
    const float* feats = (const float*)d_in[0];
    const int*   group = (const int*)  d_in[1];
    const float* w_in  = (const float*)d_in[2];
    const float* b_in  = (const float*)d_in[3];
    const float* w_out = (const float*)d_in[4];
    const float* b_out = (const float*)d_in[5];
    const float* w_cls = (const float*)d_in[6];
    const float* b_cls = (const float*)d_in[7];
    float* out = (float*)d_out;

    char* ws = (char*)d_ws;
    const size_t halfBytes = (size_t)HEADS * NROWS * DHEAD * sizeof(_Float16);
    _Float16* Qh   = (_Float16*)(ws);
    _Float16* Kh   = (_Float16*)(ws + halfBytes);
    _Float16* Vt   = (_Float16*)(ws + 2 * halfBytes);
    float*    attn = (float*)   (ws + 3 * halfBytes);
    float*    wv   = (float*)   (ws + 3 * halfBytes + (size_t)NROWS * EDIM * sizeof(float));

    qkv_proj_kernel<<<NROWS, 288, 0, stream>>>(feats, w_in, b_in, Qh, Kh, Vt);

    dim3 grid(NROWS / 16, HEADS);
    attn_kernel<<<grid, 32, 0, stream>>>(Qh, Kh, Vt, group, attn);

    fuse_cls_kernel<<<1, 128, 0, stream>>>(w_out, b_out, w_cls, b_cls, wv);
    cls_kernel<<<(NROWS + 255) / 256, 256, 0, stream>>>(attn, wv, out);
}
